// ODENet_46480136077880
// MI455X (gfx1250) — matrix operand order repaired and checked by reading, not compiled, not run on hardware
//
#include <hip/hip_runtime.h>

// ODENet forward-Euler scan on MI455X (gfx1250, wave32).
// 8 persistent workgroups x 16 independent batch columns each.
// Per step (2 barriers): VALU layer-1 tanh -> LDS f16 A tiles ->
// 4x v_wmma_f32_16x16x32_f16 per wave (W2 B-frags pinned in VGPRs) ->
// DPP8 intra-group reduce + LDS partials -> register-resident y update.

typedef __attribute__((ext_vector_type(16))) _Float16 v16h;
typedef __attribute__((ext_vector_type(8)))  _Float16 v8h;
typedef __attribute__((ext_vector_type(8)))  float    v8f;

#define T_STEPS  8192
#define BATCH    128
#define HID      128
#define BWG      16      // batch rows per workgroup (one WMMA M tile)
#define NWAVE    8
#define NTHREADS 256
#define XBLK     128     // timesteps of x staged into LDS per block
#define H1PAD    136     // 128 + 8 halfs row pad -> conflict-free b128 A loads

// DPP8 selector: per-lane source within each 8-lane group
#define DPP8_SEL(a,b,c,d,e,f,g,h) \
    ((a)|((b)<<3)|((c)<<6)|((d)<<9)|((e)<<12)|((f)<<15)|((g)<<18)|((h)<<21))
#define DPP8_XOR1 DPP8_SEL(1,0,3,2,5,4,7,6)
#define DPP8_XOR2 DPP8_SEL(2,3,0,1,6,7,4,5)
#define DPP8_XOR4 DPP8_SEL(4,5,6,7,0,1,2,3)

#define DPP8_ADD(s, SEL) \
    ((s) + __int_as_float(__builtin_amdgcn_mov_dpp8(__float_as_int(s), (SEL))))

__device__ __forceinline__ float fast_tanh(float v) {
    // tanh(v) = 1 - 2/(exp(2v)+1); v_exp_f32 + v_rcp_f32, saturates correctly.
    float e = __expf(2.0f * v);
    float r = __builtin_amdgcn_rcpf(e + 1.0f);
    return fmaf(-2.0f, r, 1.0f);
}

__global__ __launch_bounds__(NTHREADS, 1)
void odenet_scan_kernel(const float* __restrict__ x,
                        const float* __restrict__ W1,
                        const float* __restrict__ b1,
                        const float* __restrict__ W2,
                        const float* __restrict__ b2,
                        const float* __restrict__ W3,
                        const float* __restrict__ b3,
                        float* __restrict__ out)
{
    __shared__ _Float16 w2h[HID * HID];                 // 32 KB, one-time
    __shared__ __align__(16) _Float16 h1h[BWG * H1PAD]; // layer-1 output (A tiles)
    __shared__ float xbuf[XBLK][BWG];                   // 8 KB x staging
    __shared__ float w1r0[HID], w1r1[HID], b1s[HID];
    __shared__ __align__(16) float part[NWAVE][4][8];   // 8-lane-group partials

    const int tid   = threadIdx.x;
    const int b0    = blockIdx.x * BWG;
    const int wave  = tid >> 5;
    const int lane  = tid & 31;
    const int l16   = lane & 15;
    const int lhalf = lane >> 4;          // 0: lanes 0-15, 1: lanes 16-31
    const int nCol  = wave * 16 + l16;    // this lane's output column N

    // ---- one-time staging ----
    for (int j = tid; j < HID * HID; j += NTHREADS)
        w2h[j] = (_Float16)W2[j];
    if (tid < HID) {
        w1r0[tid] = W1[tid];
        w1r1[tid] = W1[HID + tid];
        b1s[tid]  = b1[tid];
    }
    if (tid < BWG) out[b0 + tid] = 0.0f;  // out[0] = y_0 = 0
    __syncthreads();

    // per-lane loop constants
    const float accInit = b2[nCol];
    const float w3n     = W3[nCol];
    const float w3n2    = -2.0f * w3n;    // tanh*W3 fused: w3n + w3n2*rcp(..)
    const float b3v     = b3[0];

    // W2 B-fragments (32x16 f16, K-chunks 0..3), pinned in VGPRs all scan.
    // B layout: lane holds column N=nCol; element e <-> K = 32c + 16*lhalf + e.
    v16h bw0, bw1, bw2, bw3;
    #pragma unroll
    for (int e = 0; e < 16; ++e) {
        const int kb = lhalf * 16 + e;
        bw0[e] = w2h[(kb +  0) * HID + nCol];
        bw1[e] = w2h[(kb + 32) * HID + nCol];
        bw2[e] = w2h[(kb + 64) * HID + nCol];
        bw3[e] = w2h[(kb + 96) * HID + nCol];
    }

    // running ODE state for this thread's batch row m = tid&15 (redundant
    // across the 16 threads sharing m; all compute identical values)
    const int   myM  = tid & 15;
    const int   myR  = myM & 7;           // part row index
    const int   myG2 = (myM >> 3) * 2;    // part group base (0 or 2)
    float ym = 0.0f;

    const int totalSteps = T_STEPS - 1;   // 8191 Euler steps
    for (int t0 = 0; t0 < totalSteps; t0 += XBLK) {
        // stage x[t0 .. t0+127][b0 .. b0+15] into LDS
        for (int j = tid; j < XBLK * BWG; j += NTHREADS) {
            const int i = j >> 4, m = j & 15;
            xbuf[i][m] = x[(size_t)(t0 + i) * BATCH + b0 + m];
        }
        // prefetch the block after this one (global_prefetch_b8)
        const int pf = t0 + XBLK;
        if (pf + XBLK <= T_STEPS && tid < XBLK)
            __builtin_prefetch(&x[(size_t)(pf + tid) * BATCH + b0], 0, 0);
        __syncthreads();                  // xbuf ready

        const int rem = totalSteps - t0;
        const int ns  = rem < XBLK ? rem : XBLK;
        #pragma unroll 1
        for (int i = 0; i < ns; ++i) {
            // ---- layer 1: h1[m][k] = tanh(x*W1[0,k] + y*W1[1,k] + b1[k]) ----
            {
                const int k0 = (tid >> 4) * 8;
                const float xm = xbuf[i][myM];
                v8h hv;
                #pragma unroll
                for (int j = 0; j < 8; ++j) {
                    const int k = k0 + j;
                    hv[j] = (_Float16)fast_tanh(
                        fmaf(xm, w1r0[k], fmaf(ym, w1r1[k], b1s[k])));
                }
                *(v8h*)&h1h[myM * H1PAD + k0] = hv;    // ds_store_b128
            }
            __syncthreads();              // B1: h1 ready, part free

            // ---- layer 2: 16x16 tile via 4 chained WMMA over K=128 ----
            v8f acc;
            #pragma unroll
            for (int r = 0; r < 8; ++r) acc[r] = accInit;

            union AFrag { v16h v; v8h h[2]; } a;
            const _Float16* arow = &h1h[l16 * H1PAD + lhalf * 16];

            a.h[0] = *(const v8h*)(arow +  0);
            a.h[1] = *(const v8h*)(arow + 8);
            acc = __builtin_amdgcn_wmma_f32_16x16x32_f16(false, a.v, false, bw0,
                                                         (short)0, acc, false, false);
            a.h[0] = *(const v8h*)(arow + 32);
            a.h[1] = *(const v8h*)(arow + 40);
            acc = __builtin_amdgcn_wmma_f32_16x16x32_f16(false, a.v, false, bw1,
                                                         (short)0, acc, false, false);
            a.h[0] = *(const v8h*)(arow + 64);
            a.h[1] = *(const v8h*)(arow + 72);
            acc = __builtin_amdgcn_wmma_f32_16x16x32_f16(false, a.v, false, bw2,
                                                         (short)0, acc, false, false);
            a.h[0] = *(const v8h*)(arow + 96);
            a.h[1] = *(const v8h*)(arow + 104);
            acc = __builtin_amdgcn_wmma_f32_16x16x32_f16(false, a.v, false, bw3,
                                                         (short)0, acc, false, false);

            // ---- layer 3: s = tanh(acc)*W3[n]; reduce across 8-lane groups
            // via DPP8 (VALU only); row M = r + 8*lhalf.
            float sred[8];
            #pragma unroll
            for (int r = 0; r < 8; ++r) {
                float e = __expf(2.0f * acc[r]);
                float rc = __builtin_amdgcn_rcpf(e + 1.0f);
                float s = fmaf(w3n2, rc, w3n);         // tanh(acc)*W3[n]
                s = DPP8_ADD(s, DPP8_XOR1);
                s = DPP8_ADD(s, DPP8_XOR2);
                s = DPP8_ADD(s, DPP8_XOR4);
                sred[r] = s;
            }
            if ((l16 & 7) == 0) {
                const int g = lhalf * 2 + (l16 >> 3);  // 8-lane group 0..3
                #pragma unroll
                for (int r = 0; r < 8; ++r) part[wave][g][r] = sred[r];
            }
            __syncthreads();              // B2: partials ready

            // ---- y update: every thread reduces its own row (register y) ----
            float s = b3v;
            #pragma unroll
            for (int w = 0; w < NWAVE; ++w)
                s += part[w][myG2][myR] + part[w][myG2 + 1][myR];
            ym += s;                      // DT = 1.0
            if (tid < BWG)
                out[(size_t)(t0 + i + 1) * BATCH + b0 + tid] = ym;
        }
    }
}

extern "C" void kernel_launch(void* const* d_in, const int* in_sizes, int n_in,
                              void* d_out, int out_size, void* d_ws, size_t ws_size,
                              hipStream_t stream) {
    const float* x  = (const float*)d_in[0];
    const float* W1 = (const float*)d_in[1];
    const float* b1 = (const float*)d_in[2];
    const float* W2 = (const float*)d_in[3];
    const float* b2 = (const float*)d_in[4];
    const float* W3 = (const float*)d_in[5];
    const float* b3 = (const float*)d_in[6];
    float* out = (float*)d_out;
    odenet_scan_kernel<<<dim3(BATCH / BWG), dim3(NTHREADS), 0, stream>>>(
        x, W1, b1, W2, b2, W3, b3, out);
}
